// AttnBlock3d_41832981463418
// MI455X (gfx1250) — matrix operand order repaired and checked by reading, not compiled, not run on hardware
//
#include <hip/hip_runtime.h>

// ---------------------------------------------------------------------------
// AttnBlock3d for MI455X (gfx1250): GroupNorm + QKV proj + flash attention +
// output proj, all matmuls on v_wmma_f32_16x16x32_bf16 (wave32).
// b=4, c=256, n=16*16*16=4096, groups=32.
// Flash-attention fusion avoids the 268MB score materialization; all WMMA
// operands are stored K-contiguous so each fragment is two global_load_b128
// straight out of the 192MB L2. Softmax runs in the exp2 domain (Q pre-scaled
// by log2(e)) so each exponential is a single v_exp_f32.
// ---------------------------------------------------------------------------

typedef __attribute__((ext_vector_type(16))) __bf16        bf16x16;
typedef __attribute__((ext_vector_type(8)))  float         f32x8;
typedef __attribute__((ext_vector_type(4)))  unsigned int  u32x4;
typedef __attribute__((ext_vector_type(4)))  float         f32x4;
typedef unsigned short bfbits;

#define CC 256
#define NN 4096
#define BB 4
#define LOG2E 1.4426950408889634f

union Pack { u32x4 u[2]; bf16x16 v; };

static __device__ __forceinline__ bfbits f2bf(float f) {
  unsigned u = __builtin_bit_cast(unsigned, f);
  u = u + 0x7FFFu + ((u >> 16) & 1u);   // round-to-nearest-even
  return (bfbits)(u >> 16);
}

// A-matrix fragment (16x32 bf16, M=row per lane%16, K contiguous in memory).
// Per ISA 7.12.2: lane<16 holds K[0..7],[16..23]; lane>=16 holds K[8..15],[24..31].
static __device__ __forceinline__ bf16x16 ld_a(const bfbits* base, long rowoff, int kstep) {
  const int lane = threadIdx.x & 31;
  const bfbits* p = base + rowoff + kstep + ((lane >> 4) << 3);
  Pack x;
  x.u[0] = *(const u32x4*)(p);
  x.u[1] = *(const u32x4*)(p + 16);
  return x.v;
}

// B-matrix fragment (32x16 bf16, N=row-of-source per lane%16, K contiguous):
// lane<16 holds K[0..7],[16..23]; lane>=16 holds K[8..15],[24..31] (matches ld_a).
static __device__ __forceinline__ bf16x16 ld_b(const bfbits* base, long rowoff, int kstep) {
  const int lane = threadIdx.x & 31;
  const bfbits* p = base + rowoff + kstep + ((lane >> 4) << 3);
  Pack x;
  x.u[0] = *(const u32x4*)(p);
  x.u[1] = *(const u32x4*)(p + 16);
  return x.v;
}

static __device__ __forceinline__ f32x8 wmma_bf16(bf16x16 a, bf16x16 b, f32x8 c) {
  return __builtin_amdgcn_wmma_f32_16x16x32_bf16(false, a, false, b, (short)0, c, false, false);
}

// ---------------------------------------------------------------------------
// Kernel: fp32 weights -> bf16 (65536 elements per matrix)
// ---------------------------------------------------------------------------
__global__ void w2bf_kernel(const float* __restrict__ w, bfbits* __restrict__ o) {
  int i = blockIdx.x * 256 + threadIdx.x;
  o[i] = f2bf(w[i]);
}

// ---------------------------------------------------------------------------
// Kernel: GroupNorm. One block per (group, batch). 8 channels * 4096 spatial
// = one contiguous 32768-float slab. Writes hnT[b][n][c] in bf16 (K-contig
// layout for the downstream WMMA B-operand).
// ---------------------------------------------------------------------------
__global__ void __launch_bounds__(256) groupnorm_kernel(
    const float* __restrict__ x, const float* __restrict__ gamma,
    const float* __restrict__ beta, bfbits* __restrict__ hnT) {
  const int g = blockIdx.x, b = blockIdx.y, t = threadIdx.x;
  const float* xg = x + ((long)b * CC + g * 8) * NN;

  float s = 0.f, sq = 0.f;
  for (int i = t * 4; i < 32768; i += 1024) {
    f32x4 v = *(const f32x4*)(xg + i);
    s  += v.x + v.y + v.z + v.w;
    sq += v.x * v.x + v.y * v.y + v.z * v.z + v.w * v.w;
  }
  __shared__ float ls[256], lq[256];
  ls[t] = s; lq[t] = sq;
  __syncthreads();
  for (int st = 128; st > 0; st >>= 1) {
    if (t < st) { ls[t] += ls[t + st]; lq[t] += lq[t + st]; }
    __syncthreads();
  }
  const float mean = ls[0] * (1.0f / 32768.0f);
  const float var  = lq[0] * (1.0f / 32768.0f) - mean * mean;
  const float rstd = rsqrtf(var + 1e-6f);

  bfbits* out = hnT + (long)b * NN * CC;
  for (int i = t * 4; i < 32768; i += 1024) {
    int c = g * 8 + (i >> 12);
    int n = i & (NN - 1);
    float ga = gamma[c], be = beta[c];
    f32x4 v = *(const f32x4*)(xg + i);
    out[(long)(n + 0) * CC + c] = f2bf((v.x - mean) * rstd * ga + be);
    out[(long)(n + 1) * CC + c] = f2bf((v.y - mean) * rstd * ga + be);
    out[(long)(n + 2) * CC + c] = f2bf((v.z - mean) * rstd * ga + be);
    out[(long)(n + 3) * CC + c] = f2bf((v.w - mean) * rstd * ga + be);
  }
}

// ---------------------------------------------------------------------------
// Kernel: C[m,n] = (A[m,:] . Bsrc[n,:] + bias[m]) * ascale  (K = 256)
// Each wave: 16(M) x 32(N) tile, 16 WMMAs, no LDS (L2-resident operands).
// mode 0: store bf16 transposed  out[n*256 + m]   (q with ascale=log2e, k)
// mode 1: store bf16 natural     out[m*4096 + n]  (v)
// mode 2: store f32 natural + residual x          (final projection)
// ---------------------------------------------------------------------------
__global__ void __launch_bounds__(256) gemm_kernel(
    const bfbits* __restrict__ A, const bfbits* __restrict__ Bsrc,
    const float* __restrict__ bias, void* __restrict__ outp,
    const float* __restrict__ xres, int mode, float ascale) {
  const int lane = threadIdx.x & 31;
  const int wv   = threadIdx.x >> 5;
  const int bz   = blockIdx.z;
  const int n0   = blockIdx.x * 256 + wv * 32;
  const int m0   = blockIdx.y * 16;

  const bfbits* B = Bsrc + (long)bz * NN * CC;
  const long arow  = (long)(m0 + (lane & 15)) * CC;
  const long brow0 = (long)(n0 + (lane & 15)) * CC;
  const long brow1 = brow0 + 16 * CC;

  f32x8 c0 = {}, c1 = {};
#pragma unroll
  for (int s = 0; s < 8; s++) {
    bf16x16 a  = ld_a(A, arow,  s * 32);
    bf16x16 b0 = ld_b(B, brow0, s * 32);
    bf16x16 b1 = ld_b(B, brow1, s * 32);
    c0 = wmma_bf16(a, b0, c0);
    c1 = wmma_bf16(a, b1, c1);
  }

#pragma unroll
  for (int r = 0; r < 8; r++) {
    const int m  = m0 + r + ((lane >> 4) << 3);
    const float bv = bias[m];
    const float v0 = (c0[r] + bv) * ascale;
    const float v1 = (c1[r] + bv) * ascale;
    const int nc0 = n0 + (lane & 15);
    const int nc1 = nc0 + 16;
    if (mode == 0) {
      bfbits* ot = (bfbits*)outp + (long)bz * NN * CC;
      ot[(long)nc0 * CC + m] = f2bf(v0);
      ot[(long)nc1 * CC + m] = f2bf(v1);
    } else if (mode == 1) {
      bfbits* on = (bfbits*)outp + (long)bz * CC * NN;
      on[(long)m * NN + nc0] = f2bf(v0);
      on[(long)m * NN + nc1] = f2bf(v1);
    } else {
      float* of = (float*)outp + (long)bz * CC * NN;
      const float* xr = xres + (long)bz * CC * NN;
      of[(long)m * NN + nc0] = v0 + xr[(long)m * NN + nc0];
      of[(long)m * NN + nc1] = v1 + xr[(long)m * NN + nc1];
    }
  }
}

// ---------------------------------------------------------------------------
// Kernel: fused flash attention. Scores arrive in exp2 domain (Q pre-scaled
// by log2e), so every exponential is one v_exp_f32. qT/kT: [n][256] bf16,
// v: [256][n] bf16, oT: [n][256] bf16.
// Block = 128 threads = 4 waves; each wave owns 16 query rows, full head dim
// (256) in registers; 64-key tiles per online-softmax round (64 WMMA/round)
// halve the rescale/shuffle overhead per key vs 32-key rounds.
// ---------------------------------------------------------------------------
__global__ void __launch_bounds__(128) attn_kernel(
    const bfbits* __restrict__ qT, const bfbits* __restrict__ kT,
    const bfbits* __restrict__ vN, bfbits* __restrict__ oT) {
  const int lane = threadIdx.x & 31;
  const int wv   = threadIdx.x >> 5;
  const int b    = blockIdx.y;
  const int i0   = blockIdx.x * 64 + wv * 16;

  const bfbits* q = qT + (long)b * NN * CC;
  const bfbits* k = kT + (long)b * NN * CC;
  const bfbits* v = vN + (long)b * CC * NN;
  bfbits*       o = oT + (long)b * NN * CC;

  __shared__ __align__(16) bfbits ldsP[4][16 * 64];
  bfbits* P = ldsP[wv];

  const long qrow = (long)(i0 + (lane & 15)) * CC;

  f32x8 acc[16];
#pragma unroll
  for (int t = 0; t < 16; t++) acc[t] = f32x8{};
  float mrow[8], lrow[8];
#pragma unroll
  for (int r = 0; r < 8; r++) { mrow[r] = -1e30f; lrow[r] = 0.f; }

  for (int kj = 0; kj < NN; kj += 64) {
    // ---- S = Q . K^T for this 16x64 tile (WMMA K over c=256) ----
    f32x8 sfr[4];
#pragma unroll
    for (int u = 0; u < 4; u++) sfr[u] = f32x8{};
    const long krow = (long)(kj + (lane & 15)) * CC;

    // prefetch next tile's K rows / V lines (speculative; safe past end)
    __builtin_prefetch(k + krow + 64 * CC, 0, 3);
    __builtin_prefetch(k + krow + 96 * CC, 0, 3);
    __builtin_prefetch(v + (long)(lane & 15) * NN + kj + 64, 0, 3);

#pragma unroll
    for (int s = 0; s < 8; s++) {
      bf16x16 a = ld_a(q, qrow, s * 32);
#pragma unroll
      for (int u = 0; u < 4; u++) {
        bf16x16 kb = ld_b(k, krow + (long)u * 16 * CC, s * 32);
        sfr[u] = wmma_bf16(a, kb, sfr[u]);
      }
    }

    // ---- online softmax over the 64 keys (exp2 domain) ----
    // C-layout: vgpr r holds row (r + 8*(lane/16)), col (lane%16).
    // xor masks 1,2,4,8 reduce within each 16-lane half (same row group).
#pragma unroll
    for (int r = 0; r < 8; r++) {
      float e0 = sfr[0][r], e1 = sfr[1][r], e2 = sfr[2][r], e3 = sfr[3][r];
      float mx = fmaxf(fmaxf(e0, e1), fmaxf(e2, e3));
      mx = fmaxf(mx, __shfl_xor(mx, 1, 32));
      mx = fmaxf(mx, __shfl_xor(mx, 2, 32));
      mx = fmaxf(mx, __shfl_xor(mx, 4, 32));
      mx = fmaxf(mx, __shfl_xor(mx, 8, 32));
      float mn = fmaxf(mrow[r], mx);
      float al = exp2f(mrow[r] - mn);
      float p0 = exp2f(e0 - mn);
      float p1 = exp2f(e1 - mn);
      float p2 = exp2f(e2 - mn);
      float p3 = exp2f(e3 - mn);
      float rs = (p0 + p1) + (p2 + p3);
      rs += __shfl_xor(rs, 1, 32);
      rs += __shfl_xor(rs, 2, 32);
      rs += __shfl_xor(rs, 4, 32);
      rs += __shfl_xor(rs, 8, 32);
      lrow[r] = lrow[r] * al + rs;
      mrow[r] = mn;
      const int row = r + ((lane >> 4) << 3);
      const int col = lane & 15;
      P[row * 64 +  0 + col] = f2bf(p0);
      P[row * 64 + 16 + col] = f2bf(p1);
      P[row * 64 + 32 + col] = f2bf(p2);
      P[row * 64 + 48 + col] = f2bf(p3);
#pragma unroll
      for (int t = 0; t < 16; t++) acc[t][r] *= al;
    }

    // ---- O += P . V^T : P(16x64) as two A-frags from wave-private LDS ----
#pragma unroll
    for (int j = 0; j < 2; j++) {
      const bfbits* pp = P + (lane & 15) * 64 + j * 32 + ((lane >> 4) << 3);
      Pack px;
      px.u[0] = *(const u32x4*)(pp);
      px.u[1] = *(const u32x4*)(pp + 16);
      bf16x16 pf = px.v;
#pragma unroll
      for (int t = 0; t < 16; t++) {
        const long vrow = (long)(t * 16 + (lane & 15)) * NN;   // row = channel c
        bf16x16 vb = ld_b(v, vrow, kj + j * 32);
        acc[t] = wmma_bf16(pf, vb, acc[t]);
      }
    }
  }

  // ---- epilogue: O /= l, store oT[n][c] bf16 ----
#pragma unroll
  for (int r = 0; r < 8; r++) {
    const float inv = 1.0f / lrow[r];
    const long orow = (long)(i0 + r + ((lane >> 4) << 3)) * CC;
#pragma unroll
    for (int t = 0; t < 16; t++)
      o[orow + t * 16 + (lane & 15)] = f2bf(acc[t][r] * inv);
  }
}

// ---------------------------------------------------------------------------
// Host-side launch. Workspace (~42.5 MB):
//   hnT | qT | kT | v | oT : each 4*4096*256 bf16, then 4 bf16 weight mats.
// ---------------------------------------------------------------------------
extern "C" void kernel_launch(void* const* d_in, const int* in_sizes, int n_in,
                              void* d_out, int out_size, void* d_ws, size_t ws_size,
                              hipStream_t stream) {
  const float* x     = (const float*)d_in[0];
  const float* gamma = (const float*)d_in[1];
  const float* beta  = (const float*)d_in[2];
  const float* wq = (const float*)d_in[3];  const float* bq = (const float*)d_in[4];
  const float* wk = (const float*)d_in[5];  const float* bk = (const float*)d_in[6];
  const float* wv = (const float*)d_in[7];  const float* bv = (const float*)d_in[8];
  const float* wo = (const float*)d_in[9];  const float* bo = (const float*)d_in[10];

  const size_t SZ = (size_t)BB * NN * CC;   // 4M elements per bf16 tensor
  bfbits* hnT = (bfbits*)d_ws;
  bfbits* qT  = hnT + SZ;
  bfbits* kT  = qT + SZ;
  bfbits* vN  = kT + SZ;
  bfbits* oT  = vN + SZ;
  bfbits* wb  = oT + SZ;                    // 4 * 65536 bf16 weights

  // 1) weights -> bf16
  w2bf_kernel<<<256, 256, 0, stream>>>(wq, wb + 0 * 65536);
  w2bf_kernel<<<256, 256, 0, stream>>>(wk, wb + 1 * 65536);
  w2bf_kernel<<<256, 256, 0, stream>>>(wv, wb + 2 * 65536);
  w2bf_kernel<<<256, 256, 0, stream>>>(wo, wb + 3 * 65536);

  // 2) GroupNorm -> hnT[b][n][c] bf16
  groupnorm_kernel<<<dim3(32, BB), 256, 0, stream>>>(x, gamma, beta, hnT);

  // 3) Q/K/V projections (WMMA GEMMs); Q pre-scaled by log2(e) for exp2 softmax
  dim3 gg(NN / 256, CC / 16, BB);
  gemm_kernel<<<gg, 256, 0, stream>>>(wb + 0 * 65536, hnT, bq, qT, nullptr, 0, LOG2E);
  gemm_kernel<<<gg, 256, 0, stream>>>(wb + 1 * 65536, hnT, bk, kT, nullptr, 0, 1.0f);
  gemm_kernel<<<gg, 256, 0, stream>>>(wb + 2 * 65536, hnT, bv, vN, nullptr, 1, 1.0f);

  // 4) fused flash attention -> oT[b][n][c]
  attn_kernel<<<dim3(NN / 64, BB), 128, 0, stream>>>(qT, kT, vN, oT);

  // 5) output projection + bias + residual -> d_out fp32
  gemm_kernel<<<gg, 256, 0, stream>>>(wb + 3 * 65536, oT, bo, d_out, x, 2, 1.0f);
}
